// ChannelAttentionRs_60370060312973
// MI455X (gfx1250) — matrix operand order repaired and checked by reading, not compiled, not run on hardware
//
#include <hip/hip_runtime.h>
#include <hip/hip_bf16.h>

// ---------------------------------------------------------------------------
// ChannelAttention for MI455X / gfx1250.
//   * qkv grouped-1x1 chains: streaming elementwise (1.2 GFLOP).
//   * Gram QK^T (K=HW) + softmax + attn@V: v_wmma_f32_16x16x32_bf16 with the
//     L2-normalization folded into Gram scaling.
//   * 3x3 conv 256->128 (38.7 GMAC, dominant): implicit GEMM, pixels = A
//     operand (ds_load_b128 from channels-last LDS halo tile, staged with
//     global_load_async_to_lds_b128), weights = B operand pre-swizzled into
//     exact per-lane fragment order (b128 from WGP$/L2), f32 WMMA
//     accumulators, fused bias + LeakyReLU + residual.
// ---------------------------------------------------------------------------

#define B_     8
#define C_     128
#define H_     128
#define W_     128
#define HW_    (H_ * W_)
#define CEXP   256
#define NHEADS 16
#define DHEAD  16

// conv tiling
#define PXT    64                    // pixels (x) per block
#define TJ     (PXT + 2)             // halo width = 66
#define CIS    264                   // LDS ci row stride (256 + 8 pad, 16B aligned)

#if defined(__has_builtin)
#if __has_builtin(__builtin_amdgcn_global_load_async_to_lds_b128)
#define HAVE_ASYNC_LDS 1
#endif
#endif

typedef __attribute__((ext_vector_type(16))) __bf16 v16bf;
typedef __attribute__((ext_vector_type(8)))  float  v8f;
typedef __attribute__((ext_vector_type(4)))  int    v4i;
typedef __attribute__((address_space(1))) v4i* as1_v4i_p;
typedef __attribute__((address_space(3))) v4i* as3_v4i_p;

union Frag16 { uint4 u[2]; v16bf v; unsigned short s[16]; };
union FragC  { v8f v; float f[8]; };

__device__ __forceinline__ unsigned short f2bf(float f) {
  unsigned u = __float_as_uint(f);
  u += 0x7FFFu + ((u >> 16) & 1u);          // round-to-nearest-even
  return (unsigned short)(u >> 16);
}
__device__ __forceinline__ float bf2f(unsigned short h) {
  return __uint_as_float(((unsigned)h) << 16);
}
__device__ __forceinline__ float lrelu(float v) { return v > 0.f ? v : 0.2f * v; }

// ---------------------------------------------------------------------------
// Kernel 1: fused grouped-1x1 chains.  q,k,v in [B][head][d][HW] bf16.
// head = e*8 + c/16, d = c%16.
// ---------------------------------------------------------------------------
__global__ __launch_bounds__(256) void qkv_kernel(
    const float* __restrict__ x,  const float* __restrict__ to,
    const float* __restrict__ w_r1, const float* __restrict__ b_r1,
    const float* __restrict__ w_r2, const float* __restrict__ b_r2,
    const float* __restrict__ w_r3, const float* __restrict__ b_r3,
    const float* __restrict__ w_t1, const float* __restrict__ b_t1,
    const float* __restrict__ w_t2, const float* __restrict__ b_t2,
    const float* __restrict__ w_t3, const float* __restrict__ b_t3,
    unsigned short* __restrict__ qo, unsigned short* __restrict__ ko,
    unsigned short* __restrict__ vo)
{
  const int c = blockIdx.y;
  const int b = blockIdx.z;
  const int p = blockIdx.x * 256 + threadIdx.x;

  const size_t in_idx = ((size_t)(b * C_ + c)) * HW_ + p;
  const float xv = x[in_idx];
  const float tv = to[in_idx];

  float t1[4];
#pragma unroll
  for (int i = 0; i < 4; ++i) t1[i] = w_r1[c * 4 + i] * xv + b_r1[c * 4 + i];
  float t2[2];
#pragma unroll
  for (int o = 0; o < 2; ++o) {
    float s = b_r2[c * 2 + o];
#pragma unroll
    for (int i = 0; i < 4; ++i) s += w_r2[(c * 2 + o) * 4 + i] * t1[i];
    t2[o] = s;
  }
  float qv[2];
#pragma unroll
  for (int o = 0; o < 2; ++o) {
    float s = b_r3[c * 2 + o];
#pragma unroll
    for (int i = 0; i < 2; ++i) s += w_r3[(c * 2 + o) * 2 + i] * t2[i];
    qv[o] = s;
  }

  float s1[4];
#pragma unroll
  for (int i = 0; i < 4; ++i) s1[i] = w_t1[c * 4 + i] * tv + b_t1[c * 4 + i];
  float s2[2];
#pragma unroll
  for (int o = 0; o < 2; ++o) {
    float s = b_t2[c * 2 + o];
#pragma unroll
    for (int i = 0; i < 4; ++i) s += w_t2[(c * 2 + o) * 4 + i] * s1[i];
    s2[o] = s;
  }
  float kv4[4];
#pragma unroll
  for (int o = 0; o < 4; ++o) {
    float s = b_t3[c * 4 + o];
#pragma unroll
    for (int i = 0; i < 2; ++i) s += w_t3[(c * 4 + o) * 2 + i] * s2[i];
    kv4[o] = s;
  }

  const int j = c & 15;
#pragma unroll
  for (int e = 0; e < 2; ++e) {
    const int h = e * 8 + (c >> 4);
    const size_t oidx = ((size_t)((b * NHEADS + h) * DHEAD + j)) * HW_ + p;
    qo[oidx] = f2bf(qv[e]);
    ko[oidx] = f2bf(kv4[e]);
    vo[oidx] = f2bf(kv4[2 + e]);
  }
}

// ---------------------------------------------------------------------------
// Kernel 2: per (b, head): Gram = Q K^T over K=HW via bf16 WMMA, fused L2
// norms + temperature + softmax, then xo = attn @ V written CHANNELS-LAST:
// xo[((b*HW + p)*256) + (d*16 + h)]  (scattered b16 stores; xo is 64MB and
// fully resident in the 192MB L2).
// ---------------------------------------------------------------------------
__global__ __launch_bounds__(256) void attn_kernel(
    const unsigned short* __restrict__ q, const unsigned short* __restrict__ k,
    const unsigned short* __restrict__ v, const float* __restrict__ temp,
    unsigned short* __restrict__ xo)
{
  __shared__ float gram[8 * 256];
  __shared__ float sqq[8 * 32];
  __shared__ float sqk[8 * 32];
  __shared__ float smat[16 * 17];
  __shared__ float nqr[16];
  __shared__ float nkr[16];

  const int h = blockIdx.x;
  const int b = blockIdx.y;
  const int tid = threadIdx.x;
  const int lane = tid & 31;
  const int wave = tid >> 5;

  const size_t head_off = ((size_t)(b * NHEADS + h)) * DHEAD * HW_;
  const unsigned short* qb = q + head_off;
  const unsigned short* kb = k + head_off;

  const int m    = lane & 15;
  const int h8   = (lane & 16) ? 8 : 0;   // A-fragment K offset
  const int boff = (lane & 16) ? 16 : 0;  // unused; B loads use h8 like A
  const unsigned short* qrow = qb + (size_t)m * HW_;
  const unsigned short* krow = kb + (size_t)m * HW_;

  FragC acc;
#pragma unroll
  for (int r = 0; r < 8; ++r) acc.f[r] = 0.f;
  float nq = 0.f, nk = 0.f;

  for (int cch = 0; cch < 64; ++cch) {
    const int p0 = (wave * 64 + cch) * 32;
    Frag16 fa, fb;
    fa.u[0] = *(const uint4*)(qrow + p0 + h8);
    fa.u[1] = *(const uint4*)(qrow + p0 + 16 + h8);
    fb.u[0] = *(const uint4*)(krow + p0 + h8);
    fb.u[1] = *(const uint4*)(krow + p0 + 16 + h8);
    acc.v = __builtin_amdgcn_wmma_f32_16x16x32_bf16(
        false, fa.v, false, fb.v, (short)0, acc.v, false, false);
#pragma unroll
    for (int i = 0; i < 16; ++i) {
      const float aq = bf2f(fa.s[i]); nq += aq * aq;
      const float ak = bf2f(fb.s[i]); nk += ak * ak;
    }
  }

#pragma unroll
  for (int r = 0; r < 8; ++r) gram[wave * 256 + r * 32 + lane] = acc.f[r];
  sqq[wave * 32 + lane] = nq;
  sqk[wave * 32 + lane] = nk;
  __syncthreads();

  if (tid < 16) {
    float aq = 0.f, ak = 0.f;
#pragma unroll
    for (int w = 0; w < 8; ++w) {
      aq += sqq[w * 32 + tid] + sqq[w * 32 + 16 + tid];
      ak += sqk[w * 32 + tid] + sqk[w * 32 + 16 + tid];
    }
    nqr[tid] = fmaxf(sqrtf(aq), 1e-12f);
    nkr[tid] = fmaxf(sqrtf(ak), 1e-12f);
  }

  const int rm = tid >> 4;
  const int cn = tid & 15;
  const int src = (rm & 7) * 32 + cn + ((rm >> 3) << 4);
  float s = 0.f;
#pragma unroll
  for (int w = 0; w < 8; ++w) s += gram[w * 256 + src];
  __syncthreads();

  const float tscale = temp[h];
  smat[rm * 17 + cn] = (s / (nqr[rm] * nkr[cn])) * tscale;
  __syncthreads();

  if (tid < 16) {
    float mx = -3.4e38f;
#pragma unroll
    for (int n = 0; n < 16; ++n) mx = fmaxf(mx, smat[tid * 17 + n]);
    float sum = 0.f;
#pragma unroll
    for (int n = 0; n < 16; ++n) {
      const float e = __expf(smat[tid * 17 + n] - mx);
      smat[tid * 17 + n] = e;
      sum += e;
    }
    const float inv = 1.f / sum;
#pragma unroll
    for (int n = 0; n < 16; ++n) smat[tid * 17 + n] *= inv;
  }
  __syncthreads();

  const unsigned short* vb = v + head_off;
  unsigned short* xb = xo + (size_t)b * HW_ * CEXP + h;
  for (int it = 0; it < 64; ++it) {
    const int p = it * 256 + tid;
    float vv[16];
#pragma unroll
    for (int e = 0; e < 16; ++e) vv[e] = bf2f(vb[(size_t)e * HW_ + p]);
    unsigned short* xp = xb + (size_t)p * CEXP;
#pragma unroll
    for (int jd = 0; jd < 16; ++jd) {
      float o = 0.f;
#pragma unroll
      for (int e = 0; e < 16; ++e) o += smat[jd * 17 + e] * vv[e];
      xp[jd * 16] = f2bf(o);
    }
  }
}

// ---------------------------------------------------------------------------
// Kernel 3: swizzle w_f [co][ci][3][3] f32 into exact WMMA B-fragment order:
// wswz[t9][cc][mt][lane][i] with co = mt*16 + (lane&15), ci = cc*32 +
// (lane&16) + i  -> each lane's fragment is one contiguous 32B run.
// ---------------------------------------------------------------------------
__global__ void wprep_kernel(const float* __restrict__ wf,
                             unsigned short* __restrict__ wswz)
{
  const int idx = blockIdx.x * 256 + threadIdx.x;
  if (idx >= 9 * 8 * 8 * 512) return;
  const int i    = idx & 15;
  const int lane = (idx >> 4) & 31;
  const int mt   = (idx >> 9) & 7;
  const int cc   = (idx >> 12) & 7;
  const int t9   = idx >> 15;
  const int co = mt * 16 + (lane & 15);
  const int ci = cc * 32 + (lane & 16) + i;
  wswz[idx] = f2bf(wf[(size_t)(co * CEXP + ci) * 9 + t9]);
}

// ---------------------------------------------------------------------------
// Kernel 4: 3x3 conv 256->128 + bias + LeakyReLU(0.2) + residual.
// Implicit GEMM, D[m=pixel, n=co].  Block = 256 thr = 8 waves; output tile =
// 128 co x 64 px of one row.  LDS halo tile [3][66][264] bf16 (104.5KB of the
// 320KB WGP LDS), staged with global_load_async_to_lds_b128 where available.
// Wave w: px-tile nt = w&3, co-tiles (w>>2)*4 + {0..3} sharing one A fragment
// (two ds_load_b128) against 4 pre-swizzled weight fragments (b128).
// ---------------------------------------------------------------------------
__global__ __launch_bounds__(256) void conv_kernel(
    const unsigned short* __restrict__ xo, const unsigned short* __restrict__ wswz,
    const float* __restrict__ bias, const float* __restrict__ xin,
    float* __restrict__ out)
{
  extern __shared__ unsigned short tile[];   // [3][66][264] bf16
  const int x0 = blockIdx.x * PXT;
  const int y  = blockIdx.y;
  const int b  = blockIdx.z;
  const int tid = threadIdx.x;

  // ---- Stage halo tile: 3*66 rows x 256 ci, 16B chunks, zero-fill borders.
  for (int idx = tid; idx < 3 * TJ * 32; idx += 256) {
    const int rowid = idx >> 5;          // 0 .. 3*66-1
    const int ck    = idx & 31;          // 16B chunk within ci
    const int r  = rowid / TJ;
    const int jj = rowid % TJ;
    const int gy = y + r - 1;
    const int gx = x0 + jj - 1;
    unsigned short* lp = &tile[(size_t)rowid * CIS + ck * 8];
    if (gy >= 0 && gy < H_ && gx >= 0 && gx < W_) {
      unsigned short* gp = const_cast<unsigned short*>(
          xo + ((size_t)(b * HW_ + gy * W_ + gx)) * CEXP + ck * 8);
#if defined(HAVE_ASYNC_LDS)
      __builtin_amdgcn_global_load_async_to_lds_b128(
          (as1_v4i_p)gp, (as3_v4i_p)lp, 0, 0);
#else
      *(uint4*)lp = *(const uint4*)gp;
#endif
    } else {
      *(uint4*)lp = make_uint4(0u, 0u, 0u, 0u);
    }
  }
#if defined(HAVE_ASYNC_LDS)
  asm volatile("s_wait_asynccnt 0" ::: "memory");
#endif
  __syncthreads();

  const int lane = tid & 31;
  const int wave = tid >> 5;
  const int nt   = wave & 3;             // pixel tile (0..3)
  const int mtg  = wave >> 2;            // co tile group (0..1)
  const int col  = lane & 15;
  const int h8   = (lane & 16) ? 8 : 0;

  FragC acc[4];
#pragma unroll
  for (int mi = 0; mi < 4; ++mi)
#pragma unroll
    for (int r = 0; r < 8; ++r) acc[mi].f[r] = 0.f;

  for (int t9 = 0; t9 < 9; ++t9) {
    const int ky = t9 / 3;
    const int kx = t9 % 3;
    const int jj = nt * 16 + col + kx;   // halo x index for this lane (row m)
    const unsigned short* trow = &tile[(size_t)(ky * TJ + jj) * CIS];
    for (int cc = 0; cc < 8; ++cc) {
      const int ci0 = cc * 32;
      Frag16 fa;
      // A = pixels [px][ci]: two contiguous 16B LDS runs per lane.
      fa.u[0] = *(const uint4*)(trow + ci0 + (lane & 16));
      fa.u[1] = *(const uint4*)(trow + ci0 + (lane & 16) + 8);
      const unsigned short* wbase =
          wswz + (size_t)(((t9 * 8 + cc) * 8 + mtg * 4) * 512) + lane * 16;
#pragma unroll
      for (int mi = 0; mi < 4; ++mi) {
        Frag16 fb;  // B = weights, pre-swizzled: one 32B run per lane.
        fb.u[0] = *(const uint4*)(wbase + mi * 512);
        fb.u[1] = *(const uint4*)(wbase + mi * 512 + 8);
        acc[mi].v = __builtin_amdgcn_wmma_f32_16x16x32_bf16(
            false, fa.v, false, fb.v, (short)0, acc[mi].v, false, false);
      }
    }
  }

  // ---- Epilogue: rows of D are consecutive x for one co -> float4 pairs.
  const int xbase = x0 + nt * 16 + ((lane >> 4) << 3);
#pragma unroll
  for (int mi = 0; mi < 4; ++mi) {
    const int co = (mtg * 4 + mi) * 16 + col;
    const float bv = bias[co];
    const size_t oi = (((size_t)b * C_ + co) * H_ + y) * W_ + xbase;
    const float4* xi = (const float4*)(xin + oi);
    const float4 ra = xi[0];
    const float4 rb = xi[1];
    float4 o0, o1;
    o0.x = ra.x + lrelu(acc[mi].f[0] + bv);
    o0.y = ra.y + lrelu(acc[mi].f[1] + bv);
    o0.z = ra.z + lrelu(acc[mi].f[2] + bv);
    o0.w = ra.w + lrelu(acc[mi].f[3] + bv);
    o1.x = rb.x + lrelu(acc[mi].f[4] + bv);
    o1.y = rb.y + lrelu(acc[mi].f[5] + bv);
    o1.z = rb.z + lrelu(acc[mi].f[6] + bv);
    o1.w = rb.w + lrelu(acc[mi].f[7] + bv);
    float4* op = (float4*)(out + oi);
    op[0] = o0;
    op[1] = o1;
  }
}

// ---------------------------------------------------------------------------
extern "C" void kernel_launch(void* const* d_in, const int* in_sizes, int n_in,
                              void* d_out, int out_size, void* d_ws, size_t ws_size,
                              hipStream_t stream) {
  (void)in_sizes; (void)n_in; (void)out_size; (void)ws_size;
  const float* x    = (const float*)d_in[0];
  const float* to   = (const float*)d_in[1];
  const float* t    = (const float*)d_in[2];
  const float* w_r1 = (const float*)d_in[3];
  const float* b_r1 = (const float*)d_in[4];
  const float* w_r2 = (const float*)d_in[5];
  const float* b_r2 = (const float*)d_in[6];
  const float* w_r3 = (const float*)d_in[7];
  const float* b_r3 = (const float*)d_in[8];
  const float* w_t1 = (const float*)d_in[9];
  const float* b_t1 = (const float*)d_in[10];
  const float* w_t2 = (const float*)d_in[11];
  const float* b_t2 = (const float*)d_in[12];
  const float* w_t3 = (const float*)d_in[13];
  const float* b_t3 = (const float*)d_in[14];
  const float* w_f  = (const float*)d_in[15];
  const float* b_f  = (const float*)d_in[16];
  float* out = (float*)d_out;

  const size_t per = (size_t)B_ * CEXP * HW_;     // bf16 elements
  unsigned short* qb   = (unsigned short*)d_ws;
  unsigned short* kb   = qb + per;
  unsigned short* vb   = kb + per;
  unsigned short* xob  = vb + per;                // channels-last
  unsigned short* wszb = xob + per;

  qkv_kernel<<<dim3(HW_ / 256, C_, B_), 256, 0, stream>>>(
      x, to, w_r1, b_r1, w_r2, b_r2, w_r3, b_r3,
      w_t1, b_t1, w_t2, b_t2, w_t3, b_t3, qb, kb, vb);

  wprep_kernel<<<dim3((9 * 8 * 8 * 512 + 255) / 256), 256, 0, stream>>>(w_f, wszb);

  attn_kernel<<<dim3(NHEADS, B_), 256, 0, stream>>>(qb, kb, vb, t, xob);

  conv_kernel<<<dim3(W_ / PXT, H_, B_), 256,
                (size_t)3 * TJ * CIS * sizeof(unsigned short), stream>>>(
      xob, wszb, b_f, x, out);
}
